// EfficientMultiHeadWSA_42485816492342
// MI455X (gfx1250) — matrix operand order repaired and checked by reading, not compiled, not run on hardware
//
#include <hip/hip_runtime.h>
#include <hip/hip_bf16.h>

typedef __attribute__((ext_vector_type(16))) _Float16 v16h;
typedef __attribute__((ext_vector_type(8)))  _Float16 v8h;
typedef __attribute__((ext_vector_type(8)))  float    v8f;

#define EDIM   256
#define SA     264      // stride (halves) of 64x256 f16 A-tile in LDS
#define SQ     776      // stride (halves) of 64x768 f16 qkv in LDS
#define SF     260      // stride (floats) of 64x256 f32 final tile (aliases s_q)
#define WSCALE 0.17677669529663687f   // 32^-0.5

__device__ __forceinline__ v16h cat16(v8h lo, v8h hi) {
    return __builtin_shufflevector(lo, hi, 0,1,2,3,4,5,6,7,8,9,10,11,12,13,14,15);
}

// A fragment (16x32 f16) from an LDS tile, row-major with stride SA halves.
// Lanes 0-15: row=lane, K 0-7 then 16-23 ; lanes 16-31: same rows, K 8-15 then 24-31.
__device__ __forceinline__ v16h load_a_frag(const _Float16* base, int mblk, int kblk, int lane) {
    const int row  = lane & 15;
    const int half = lane >> 4;
    const _Float16* p = base + (mblk * 16 + row) * SA + kblk * 32;
    v8h lo = *(const v8h*)(p + half * 8);
    v8h hi = *(const v8h*)(p + 16 + half * 8);
    return cat16(lo, hi);
}

// B fragment (32x16 f16) straight from global f16 weights, row-major [Nout][256].
// B[k][n] = W[n][k]; lane%16 = column n; lanes 0-15: K 0-7 then 16-23, lanes 16-31: K 8-15 then 24-31.
__device__ __forceinline__ v16h load_b_frag(const _Float16* w, int nblk, int kblk, int lane) {
    const int col  = lane & 15;
    const int half = lane >> 4;
    const _Float16* p = w + (nblk * 16 + col) * EDIM + kblk * 32;
    v8h lo = *(const v8h*)(p + half * 8);
    v8h hi = *(const v8h*)(p + 16 + half * 8);
    return cat16(lo, hi);
}

// ---------------------------------------------------------------------------
// Prep: convert fp32 weights to f16 in workspace (weights are L2-resident).
// ---------------------------------------------------------------------------
__global__ void wsa_prep_kernel(const float* __restrict__ wqkv,
                                const float* __restrict__ wout,
                                _Float16* __restrict__ wqkv_h,
                                _Float16* __restrict__ wout_h) {
    const int i = blockIdx.x * 256 + threadIdx.x;
    if (i < 768 * 256) wqkv_h[i] = (_Float16)wqkv[i];
    if (i < 256 * 256) wout_h[i] = (_Float16)wout[i];
}

// ---------------------------------------------------------------------------
// Fused window attention: one workgroup (8 waves) per 8x8 window.
// ---------------------------------------------------------------------------
__global__ void __launch_bounds__(256)
wsa_main_kernel(const float* __restrict__ x,          // (8, 256, 128, 128)
                const _Float16* __restrict__ wqkv_h,  // (768, 256) f16
                const _Float16* __restrict__ wout_h,  // (256, 256) f16
                const float* __restrict__ b_out,      // (256,)
                float* __restrict__ out) {            // (8, 256, 128, 128) = (B,C,W,H)
    __shared__ __align__(16) _Float16 s_a[64 * SA];   // 33.8 KB: xw tile, later out2 tile
    __shared__ __align__(16) _Float16 s_q[64 * SQ];   // 99.3 KB: qkv f16, later final f32

    const int w    = blockIdx.x;          // window id 0..2047
    const int b    = w >> 8;
    const int widx = w & 255;
    const int wh   = widx >> 4;           // window row (H blocks)
    const int ww   = widx & 15;           // window col (W blocks)
    const int tid  = threadIdx.x;
    const int ws   = tid >> 5;            // wave id 0..7
    const int lane = tid & 31;

    // ---------------- Stage 0: gather window, fp32 -> f16, into s_a ----------
    // x[b][ch][wh*8+r][ww*8 .. +7] : 32B contiguous runs.
    #pragma unroll
    for (int it = 0; it < 8; ++it) {
        const int idx = it * 256 + tid;   // 0..2047
        const int ch  = idx >> 3;
        const int r   = idx & 7;
        const float* xp = x + (((long)(b * EDIM + ch) * 128) + wh * 8 + r) * 128 + ww * 8;
        float4 a0 = *(const float4*)xp;
        float4 a1 = *((const float4*)xp + 1);
        _Float16* dp = s_a + (r * 8) * SA + ch;
        dp[0 * SA] = (_Float16)a0.x;  dp[1 * SA] = (_Float16)a0.y;
        dp[2 * SA] = (_Float16)a0.z;  dp[3 * SA] = (_Float16)a0.w;
        dp[4 * SA] = (_Float16)a1.x;  dp[5 * SA] = (_Float16)a1.y;
        dp[6 * SA] = (_Float16)a1.z;  dp[7 * SA] = (_Float16)a1.w;
    }
    __syncthreads();

    // ---------------- Stage 1: QKV = xw @ w_qkv^T   (64x256 @ 256x768) -------
    // 4 mblk x 48 nblk tiles; wave ws owns nblk = ws*6 .. ws*6+5.
    {
        const int col = lane & 15;
        #pragma unroll
        for (int j = 0; j < 6; ++j) {
            const int nblk = ws * 6 + j;
            v16h bf[8];
            #pragma unroll
            for (int kb = 0; kb < 8; ++kb)
                bf[kb] = load_b_frag(wqkv_h, nblk, kb, lane);
            #pragma unroll
            for (int mb = 0; mb < 4; ++mb) {
                v8f acc = {};
                #pragma unroll
                for (int kb = 0; kb < 8; ++kb) {
                    v16h af = load_a_frag(s_a, mb, kb, lane);
                    acc = __builtin_amdgcn_wmma_f32_16x16x32_f16(
                        false, af, false, bf[kb], (short)0, acc, false, false);
                }
                const int rbase = mb * 16 + (lane >> 4) * 8;
                _Float16* qp = s_q + nblk * 16 + col;
                #pragma unroll
                for (int r = 0; r < 8; ++r)
                    qp[(rbase + r) * SQ] = (_Float16)acc[r];
            }
        }
    }
    __syncthreads();

    // ---------------- Stage 2: per-token cross-head attention ---------------
    // attn[n,h,g] = softmax_g( Q[n,h,:]·K[n,g,:] * SCALE );  O = attn @ V.
    // Write permuted: out2[h*8 + n/8][(n&7)*32 + d]  (into s_a, reused).
    {
        const int n  = tid >> 2;          // token 0..63
        const int hp = tid & 3;           // 2 heads per thread
        const _Float16* kp = s_q + n * SQ + 256;
        const _Float16* vp = s_q + n * SQ + 512;
        #pragma unroll
        for (int hh = 0; hh < 2; ++hh) {
            const int h = hp * 2 + hh;
            const _Float16* qp = s_q + n * SQ + h * 32;
            float q[32];
            #pragma unroll
            for (int d = 0; d < 32; ++d) q[d] = (float)qp[d];
            float s[8];
            float m = -1e30f;
            #pragma unroll
            for (int g = 0; g < 8; ++g) {
                float acc = 0.f;
                #pragma unroll
                for (int d = 0; d < 32; ++d) acc += q[d] * (float)kp[g * 32 + d];
                s[g] = acc * WSCALE;
                m = fmaxf(m, s[g]);
            }
            float p[8], sum = 0.f;
            #pragma unroll
            for (int g = 0; g < 8; ++g) { p[g] = __expf(s[g] - m); sum += p[g]; }
            const float inv = 1.f / sum;
            _Float16* op = s_a + (h * 8 + (n >> 3)) * SA + (n & 7) * 32;
            #pragma unroll
            for (int d = 0; d < 32; ++d) {
                float o = 0.f;
                #pragma unroll
                for (int g = 0; g < 8; ++g) o += p[g] * (float)vp[g * 32 + d];
                op[d] = (_Float16)(o * inv);
            }
        }
    }
    __syncthreads();

    // ---------------- Stage 3: final = out2 @ w_out^T  (64x256 @ 256x256) ----
    // Result f32 goes to s_f (aliases s_q region, qkv data is dead).
    float* s_f = (float*)s_q;
    {
        const int col = lane & 15;
        #pragma unroll
        for (int j = 0; j < 2; ++j) {
            const int nblk = ws * 2 + j;
            v16h bf[8];
            #pragma unroll
            for (int kb = 0; kb < 8; ++kb)
                bf[kb] = load_b_frag(wout_h, nblk, kb, lane);
            #pragma unroll
            for (int mb = 0; mb < 4; ++mb) {
                v8f acc = {};
                #pragma unroll
                for (int kb = 0; kb < 8; ++kb) {
                    v16h af = load_a_frag(s_a, mb, kb, lane);
                    acc = __builtin_amdgcn_wmma_f32_16x16x32_f16(
                        false, af, false, bf[kb], (short)0, acc, false, false);
                }
                const int rbase = mb * 16 + (lane >> 4) * 8;
                float* fp = s_f + nblk * 16 + col;
                #pragma unroll
                for (int r = 0; r < 8; ++r)
                    fp[(rbase + r) * SF] = acc[r];
            }
        }
    }
    __syncthreads();

    // ---------------- Stage 4: bias + write (B, C, W, H) layout --------------
    // out[b][ch][ww*8+pc][wh*8 + 0..7] : 32B contiguous along h.
    #pragma unroll
    for (int it = 0; it < 8; ++it) {
        const int idx = it * 256 + tid;   // 0..2047
        const int ch  = idx >> 3;
        const int pc  = idx & 7;
        const float bias = b_out[ch];
        float v[8];
        #pragma unroll
        for (int pr = 0; pr < 8; ++pr)
            v[pr] = s_f[(pr * 8 + pc) * SF + ch] + bias;
        float* op = out + (((long)(b * EDIM + ch) * 128) + ww * 8 + pc) * 128 + wh * 8;
        float4 v0 = {v[0], v[1], v[2], v[3]};
        float4 v1 = {v[4], v[5], v[6], v[7]};
        *(float4*)op       = v0;
        *((float4*)op + 1) = v1;
    }
}

// ---------------------------------------------------------------------------
extern "C" void kernel_launch(void* const* d_in, const int* in_sizes, int n_in,
                              void* d_out, int out_size, void* d_ws, size_t ws_size,
                              hipStream_t stream) {
    const float* x     = (const float*)d_in[0];   // 8*256*128*128
    const float* wqkv  = (const float*)d_in[1];   // 768*256
    const float* wout  = (const float*)d_in[2];   // 256*256
    const float* bout  = (const float*)d_in[3];   // 256
    float* out = (float*)d_out;

    _Float16* wqkv_h = (_Float16*)d_ws;                                  // 393,216 B
    _Float16* wout_h = (_Float16*)((char*)d_ws + 768 * 256 * sizeof(_Float16)); // +131,072 B

    wsa_prep_kernel<<<768, 256, 0, stream>>>(wqkv, wout, wqkv_h, wout_h);
    wsa_main_kernel<<<2048, 256, 0, stream>>>(x, wqkv_h, wout_h, bout, out);
}
